// NeuralODEClassifier_14096082666134
// MI455X (gfx1250) — compile-verified, matrix operand order checked by reading, not run on hardware
//
#include <hip/hip_runtime.h>
#include <cmath>

typedef __attribute__((ext_vector_type(16))) _Float16 v16h;
typedef __attribute__((ext_vector_type(8)))  _Float16 v8h;
typedef __attribute__((ext_vector_type(4)))  _Float16 v4h;
typedef __attribute__((ext_vector_type(8)))  float    v8f;

#define WMMA(a, b, c) \
  __builtin_amdgcn_wmma_f32_16x16x32_f16(false, (a), false, (b), (short)0, (c), false, false)

constexpr int Dv = 784;
constexpr int Hv = 64;
constexpr int KB1 = 25;               // ceil(784/32) K-blocks for input proj
constexpr int BLOCK_ROWS = 128;       // 8 waves * 16 rows
constexpr int THREADS = 256;

// LDS partition (units: _Float16 elements)
constexpr int SZ_WINF = KB1 * 4 * 32 * 16;   // W_in B-fragments (51200)
constexpr int SZ_WF   = 4 * 2 * 32 * 16;     // W1 / W2 fragments (4096 each)
constexpr int SZ_X    = BLOCK_ROWS * 32;     // one x staging buffer (4096)
constexpr int SZ_H    = BLOCK_ROWS * 64;     // h staging (C->A bounce)
constexpr int SZ_ALL  = SZ_WINF + 2 * SZ_WF + 2 * SZ_X + SZ_H;   // 75776 halves

__device__ __forceinline__ float fast_tanh(float v) {
#if __has_builtin(__builtin_amdgcn_tanhf)
  return __builtin_amdgcn_tanhf(v);     // v_tanh_f32 (CDNA5 trans unit)
#else
  return tanhf(v);
#endif
}

// A-fragment (16-bit 16x32): lane L holds row m=L&15; half=L>>4 selects
// K sub-ranges {8h..8h+7} and {16+8h..16+8h+7} (two contiguous 16B chunks).
__device__ __forceinline__ v16h load_a_frag(const _Float16* rowp, int half) {
  v8h lo = *(const v8h*)(rowp + 8 * half);
  v8h hi = *(const v8h*)(rowp + 16 + 8 * half);
  v16h a;
#pragma unroll
  for (int i = 0; i < 8; ++i) { a[i] = lo[i]; a[8 + i] = hi[i]; }
  return a;
}

// f(h) = tanh(h@W1+b1)@W2+b2 on one wave's 16x64 tile held in C-fragments.
__device__ __forceinline__ void feval(const v8f hin[4], v8f kout[4],
                                      _Float16* sHw,
                                      const _Float16* sW1f, const _Float16* sW2f,
                                      const float* b1v, const float* b2v,
                                      int lane, int m, int half) {
#pragma unroll
  for (int nb = 0; nb < 4; ++nb) {
    int n = nb * 16 + m;
#pragma unroll
    for (int r = 0; r < 8; ++r)
      sHw[(r + 8 * half) * 64 + n] = (_Float16)hin[nb][r];
  }
  asm volatile("s_wait_dscnt 0" ::: "memory");   // wave-local LDS RAW fence

  const _Float16* rowp = sHw + m * 64;
  v16h a0 = load_a_frag(rowp, half);             // K 0..31
  v16h a1 = load_a_frag(rowp + 32, half);        // K 32..63

  v8f t[4];
#pragma unroll
  for (int nb = 0; nb < 4; ++nb) {
    v8f c;
#pragma unroll
    for (int r = 0; r < 8; ++r) c[r] = b1v[nb];
    v16h bw0 = *(const v16h*)(sW1f + ((nb * 2 + 0) * 32 + lane) * 16);
    v16h bw1 = *(const v16h*)(sW1f + ((nb * 2 + 1) * 32 + lane) * 16);
    c = WMMA(a0, bw0, c);
    c = WMMA(a1, bw1, c);
#pragma unroll
    for (int r = 0; r < 8; ++r) t[nb][r] = fast_tanh(c[r]);
  }

#pragma unroll
  for (int nb = 0; nb < 4; ++nb) {
    int n = nb * 16 + m;
#pragma unroll
    for (int r = 0; r < 8; ++r)
      sHw[(r + 8 * half) * 64 + n] = (_Float16)t[nb][r];
  }
  asm volatile("s_wait_dscnt 0" ::: "memory");

  v16h u0 = load_a_frag(rowp, half);
  v16h u1 = load_a_frag(rowp + 32, half);
#pragma unroll
  for (int nb = 0; nb < 4; ++nb) {
    v8f c;
#pragma unroll
    for (int r = 0; r < 8; ++r) c[r] = b2v[nb];
    v16h bw0 = *(const v16h*)(sW2f + ((nb * 2 + 0) * 32 + lane) * 16);
    v16h bw1 = *(const v16h*)(sW2f + ((nb * 2 + 1) * 32 + lane) * 16);
    c = WMMA(u0, bw0, c);
    c = WMMA(u1, bw1, c);
    kout[nb] = c;
  }
}

__global__ void __launch_bounds__(THREADS, 1)
node_megakernel(const float* __restrict__ x,
                const float* __restrict__ W_in, const float* __restrict__ b_in,
                const float* __restrict__ gamma1, const float* __restrict__ beta1,
                const float* __restrict__ W1, const float* __restrict__ b1,
                const float* __restrict__ W2, const float* __restrict__ b2,
                const float* __restrict__ gamma2, const float* __restrict__ beta2,
                const float* __restrict__ W_out, const float* __restrict__ b_out,
                const int* __restrict__ num_steps_p,
                float* __restrict__ out, int Btot) {
  extern __shared__ ulong4 smem4[];
  _Float16* sWinF = (_Float16*)smem4;
  _Float16* sW1f  = sWinF + SZ_WINF;
  _Float16* sW2f  = sW1f + SZ_WF;
  _Float16* sX0   = sW2f + SZ_WF;
  _Float16* sX1   = sX0 + SZ_X;
  _Float16* sH    = sX1 + SZ_X;
  float*    sEp   = (float*)smem4;   // reuses W_in region after stage 1

  const int tid  = threadIdx.x;
  const int lane = tid & 31;
  const int wave = tid >> 5;
  const int m    = lane & 15;
  const int half = lane >> 4;
  const long row0 = (long)blockIdx.x * BLOCK_ROWS;

  // ---------- W_in -> LDS B-fragments ----------
  // Coalesced float4 reads over flat j = k*64+n (k < 784), batched 4-wide so
  // four global loads are in flight before the single wait; scatter to LDS.
  {
    constexpr int NW4 = (Dv * Hv) / 4;         // 12544 float4 chunks
    constexpr int ITERS = (NW4 + THREADS - 1) / THREADS;  // 49
#pragma unroll
    for (int it = 0; it < ITERS; it += 4) {
      float4 w[4];
#pragma unroll
      for (int u = 0; u < 4; ++u) {
        int itu = (it + u < ITERS) ? (it + u) : (ITERS - 1);   // clamp (compile-time)
        w[u] = ((const float4*)W_in)[itu * THREADS + tid];
      }
#pragma unroll
      for (int u = 0; u < 4; ++u) {
        if (it + u < ITERS) {
          int j = (it + u) * THREADS * 4 + tid * 4;
          int k = j >> 6, n = j & 63;
          int kb = k >> 5, kl = k & 31;
          int hl = (kl >> 3) & 1;
          int e  = (kl & 7) + 8 * (kl >> 4);
          int base = ((kb * 4 + (n >> 4)) * 32 + 16 * hl + (n & 15)) * 16 + e;
          sWinF[base +  0] = (_Float16)w[u].x;
          sWinF[base + 16] = (_Float16)w[u].y;
          sWinF[base + 32] = (_Float16)w[u].z;
          sWinF[base + 48] = (_Float16)w[u].w;
        }
      }
    }
    // zero-pad fragments for k in [784, 800): kb=24, e in 8..15, all lanes
#pragma unroll
    for (int i = tid; i < 4 * 32 * 8; i += THREADS) {
      int e = 8 + (i & 7), ln = (i >> 3) & 31, nb = i >> 8;
      sWinF[((24 * 4 + nb) * 32 + ln) * 16 + e] = (_Float16)0.0f;
    }
  }
  // ---------- W1 / W2 -> LDS B-fragments (two-phase, 8 loads in flight) ----
  {
    float4 wa[4], wb[4];
#pragma unroll
    for (int u = 0; u < 4; ++u) {
      int j4 = u * THREADS + tid;                 // 1024 chunks total
      wa[u] = ((const float4*)W1)[j4];
      wb[u] = ((const float4*)W2)[j4];
    }
#pragma unroll
    for (int u = 0; u < 4; ++u) {
      int j = (u * THREADS + tid) * 4;
      int k = j >> 6, n = j & 63;
      int kb = k >> 5, kl = k & 31;
      int hl = (kl >> 3) & 1;
      int e  = (kl & 7) + 8 * (kl >> 4);
      int base = (((n >> 4) * 2 + kb) * 32 + 16 * hl + (n & 15)) * 16 + e;
      sW1f[base +  0] = (_Float16)wa[u].x;
      sW1f[base + 16] = (_Float16)wa[u].y;
      sW1f[base + 32] = (_Float16)wa[u].z;
      sW1f[base + 48] = (_Float16)wa[u].w;
      sW2f[base +  0] = (_Float16)wb[u].x;
      sW2f[base + 16] = (_Float16)wb[u].y;
      sW2f[base + 32] = (_Float16)wb[u].z;
      sW2f[base + 48] = (_Float16)wb[u].w;
    }
  }

  // per-lane channel constants (n = nb*16 + m)
  v8f acc[4];
  float b1v[4], b2v[4], g1v[4], be1v[4], g2v[4], be2v[4];
#pragma unroll
  for (int nb = 0; nb < 4; ++nb) {
    int n = nb * 16 + m;
    float bi = b_in[n];
#pragma unroll
    for (int r = 0; r < 8; ++r) acc[nb][r] = bi;
    b1v[nb] = b1[n];     b2v[nb] = b2[n];
    g1v[nb] = gamma1[n]; be1v[nb] = beta1[n];
    g2v[nb] = gamma2[n]; be2v[nb] = beta2[n];
  }

  // ---------- x tile: unconditional clamped loads + masked convert ----------
  const int r_   = (tid * 4) >> 5;               // row handled per chunk base
  // chunk idx = tid + jj*THREADS; row = idx>>3, kq = idx&7
  auto load_x_chunks = [&](int kb, float4* v, bool* ok) {
#pragma unroll
    for (int jj = 0; jj < 4; ++jj) {
      int idx = tid + jj * THREADS;
      int r  = idx >> 3;
      int kq = idx & 7;
      long gr = row0 + r;
      bool rok = (gr < (long)Btot);
      if (!rok) gr = (long)Btot - 1;
      int gk = kb * 32 + kq * 4;
      bool kok = (gk + 4 <= Dv);
      int gkc = kok ? gk : (Dv - 4);
      ok[jj] = rok && kok;
      v[jj] = *(const float4*)(x + gr * (long)Dv + gkc);   // always in-bounds
    }
  };
  auto store_x_chunks = [&](const float4* v, const bool* ok, _Float16* buf) {
#pragma unroll
    for (int jj = 0; jj < 4; ++jj) {
      int idx = tid + jj * THREADS;
      int r  = idx >> 3;
      int kq = idx & 7;
      float4 w = v[jj];
      float zm = ok[jj] ? 1.0f : 0.0f;
      v4h hv = { (_Float16)(w.x * zm), (_Float16)(w.y * zm),
                 (_Float16)(w.z * zm), (_Float16)(w.w * zm) };
      *(v4h*)(buf + r * 32 + kq * 4) = hv;
    }
  };
  (void)r_;

  float4 pref[4];
  bool   pok[4];
  load_x_chunks(0, pref, pok);
  store_x_chunks(pref, pok, sX0);
  __syncthreads();

  // --- stage 1: h0_pre = x @ W_in + b_in, software-pipelined ---
  for (int kb = 0; kb < KB1; ++kb) {
    _Float16* cur = (kb & 1) ? sX1 : sX0;
    _Float16* nxt = (kb & 1) ? sX0 : sX1;
    if (kb + 1 < KB1) load_x_chunks(kb + 1, pref, pok);   // issue global loads
    // WMMAs for current block run while the loads are in flight
    v16h a = load_a_frag(cur + (wave * 16 + m) * 32, half);
#pragma unroll
    for (int nb = 0; nb < 4; ++nb) {
      v16h b = *(const v16h*)(sWinF + ((kb * 4 + nb) * 32 + lane) * 16);
      acc[nb] = WMMA(a, b, acc[nb]);
    }
    if (kb + 1 < KB1) store_x_chunks(pref, pok, nxt);     // wait + convert + ds
    __syncthreads();
  }

  // --- GroupNorm(8) + tanh: group = 8 consecutive lanes in C layout ---
  v8f h[4];
#pragma unroll
  for (int nb = 0; nb < 4; ++nb) {
#pragma unroll
    for (int r = 0; r < 8; ++r) {
      float v = acc[nb][r];
      float s = v, s2 = v * v;
      s += __shfl_xor(s, 1, 32);  s2 += __shfl_xor(s2, 1, 32);
      s += __shfl_xor(s, 2, 32);  s2 += __shfl_xor(s2, 2, 32);
      s += __shfl_xor(s, 4, 32);  s2 += __shfl_xor(s2, 4, 32);
      float mu  = s * 0.125f;
      float var = s2 * 0.125f - mu * mu;
      float inv = rsqrtf(var + 1e-5f);
      h[nb][r] = fast_tanh((v - mu) * inv * g1v[nb] + be1v[nb]);
    }
  }

  // --- RK4 integration, h resident in registers ---
  const int nstep = *num_steps_p;
  const float dt = 1.0f / (float)nstep;
  _Float16* sHw = sH + wave * 16 * 64;

  v8f k[4], ksum[4], harg[4];
  for (int s = 0; s < nstep; ++s) {
    feval(h, k, sHw, sW1f, sW2f, b1v, b2v, lane, m, half);           // k1
#pragma unroll
    for (int nb = 0; nb < 4; ++nb)
#pragma unroll
      for (int r = 0; r < 8; ++r) {
        ksum[nb][r] = k[nb][r];
        harg[nb][r] = h[nb][r] + 0.5f * dt * k[nb][r];
      }
    feval(harg, k, sHw, sW1f, sW2f, b1v, b2v, lane, m, half);        // k2
#pragma unroll
    for (int nb = 0; nb < 4; ++nb)
#pragma unroll
      for (int r = 0; r < 8; ++r) {
        ksum[nb][r] += 2.0f * k[nb][r];
        harg[nb][r] = h[nb][r] + 0.5f * dt * k[nb][r];
      }
    feval(harg, k, sHw, sW1f, sW2f, b1v, b2v, lane, m, half);        // k3
#pragma unroll
    for (int nb = 0; nb < 4; ++nb)
#pragma unroll
      for (int r = 0; r < 8; ++r) {
        ksum[nb][r] += 2.0f * k[nb][r];
        harg[nb][r] = h[nb][r] + dt * k[nb][r];
      }
    feval(harg, k, sHw, sW1f, sW2f, b1v, b2v, lane, m, half);        // k4
#pragma unroll
    for (int nb = 0; nb < 4; ++nb)
#pragma unroll
      for (int r = 0; r < 8; ++r) {
        ksum[nb][r] += k[nb][r];
        h[nb][r] += (dt / 6.0f) * ksum[nb][r];
      }
  }

  // --- epilogue: GroupNorm2 -> stage rows (stride 65, bank-conflict-free) ---
#pragma unroll
  for (int nb = 0; nb < 4; ++nb) {
#pragma unroll
    for (int r = 0; r < 8; ++r) {
      float v = h[nb][r];
      float s = v, s2 = v * v;
      s += __shfl_xor(s, 1, 32);  s2 += __shfl_xor(s2, 1, 32);
      s += __shfl_xor(s, 2, 32);  s2 += __shfl_xor(s2, 2, 32);
      s += __shfl_xor(s, 4, 32);  s2 += __shfl_xor(s2, 4, 32);
      float mu  = s * 0.125f;
      float var = s2 * 0.125f - mu * mu;
      float inv = rsqrtf(var + 1e-5f);
      float nv  = (v - mu) * inv * g2v[nb] + be2v[nb];
      sEp[(wave * 16 + r + 8 * half) * 65 + nb * 16 + m] = nv;
    }
  }
  __syncthreads();

  // tiny 64->10 head + log_softmax, one row per lane
  if (tid < BLOCK_ROWS) {
    long grow = row0 + tid;
    if (grow < Btot) {
      const float* hr = sEp + tid * 65;
      float logit[10];
#pragma unroll
      for (int j = 0; j < 10; ++j) {
        float a = b_out[j];
#pragma unroll
        for (int c = 0; c < 64; ++c) a = fmaf(hr[c], W_out[c * 10 + j], a);
        logit[j] = a;
      }
      float mx = logit[0];
#pragma unroll
      for (int j = 1; j < 10; ++j) mx = fmaxf(mx, logit[j]);
      float se = 0.0f;
#pragma unroll
      for (int j = 0; j < 10; ++j) se += __expf(logit[j] - mx);
      float lse = mx + __logf(se);
#pragma unroll
      for (int j = 0; j < 10; ++j) out[grow * 10 + j] = logit[j] - lse;
    }
  }
}

extern "C" void kernel_launch(void* const* d_in, const int* in_sizes, int n_in,
                              void* d_out, int out_size, void* d_ws, size_t ws_size,
                              hipStream_t stream) {
  const float* x      = (const float*)d_in[0];
  const float* W_in   = (const float*)d_in[1];
  const float* b_in   = (const float*)d_in[2];
  const float* gamma1 = (const float*)d_in[3];
  const float* beta1  = (const float*)d_in[4];
  const float* W1     = (const float*)d_in[5];
  const float* b1     = (const float*)d_in[6];
  const float* W2     = (const float*)d_in[7];
  const float* b2     = (const float*)d_in[8];
  const float* gamma2 = (const float*)d_in[9];
  const float* beta2  = (const float*)d_in[10];
  const float* W_out  = (const float*)d_in[11];
  const float* b_out  = (const float*)d_in[12];
  const int*   nstep  = (const int*)d_in[13];

  int Btot = in_sizes[0] / Dv;
  int grid = (Btot + BLOCK_ROWS - 1) / BLOCK_ROWS;
  size_t shmem = (size_t)SZ_ALL * sizeof(_Float16);

  hipFuncSetAttribute((const void*)node_megakernel,
                      hipFuncAttributeMaxDynamicSharedMemorySize, (int)shmem);

  node_megakernel<<<grid, THREADS, shmem, stream>>>(
      x, W_in, b_in, gamma1, beta1, W1, b1, W2, b2, gamma2, beta2,
      W_out, b_out, nstep, (float*)d_out, Btot);
}
